// LRULayer_67980742362003
// MI455X (gfx1250) — matrix operand order repaired and checked by reading, not compiled, not run on hardware
//
#include <hip/hip_runtime.h>
#include <hip/hip_bf16.h>
#include <stdint.h>

// ---------------------------------------------------------------------------
// LRU block for MI455X (gfx1250, wave32, WMMA).
// GEMMs: 128x128 block tile, 8 waves (2x4), 8 v_wmma_f32_16x16x32_f16 per
// K-step per wave, double-buffered LDS fed by global_load_async_to_lds_b128
// (ASYNCcnt) so copy of tile i+1 overlaps compute of tile i.
// Weights are pre-transposed to [N,K] f16 so both tiles load contiguously.
// ---------------------------------------------------------------------------

typedef __attribute__((ext_vector_type(16))) _Float16 v16h;
typedef __attribute__((ext_vector_type(8)))  float    v8f;
typedef __attribute__((ext_vector_type(4)))  unsigned int u32x4;

#define LRU_B   16
#define LRU_T   2048
#define LRU_D   1024
#define GEMM_TM 128
#define GEMM_TN 128
#define GEMM_TK 32

// --- CDNA5 async copy: global -> LDS, 16 bytes per lane, tracked by ASYNCcnt
__device__ __forceinline__ void async_copy_b128(unsigned lds_addr,
                                                const _Float16* gptr) {
  asm volatile("global_load_async_to_lds_b128 %0, %1, off"
               :: "v"(lds_addr), "v"(gptr)
               : "memory");
}
__device__ __forceinline__ void wait_asynccnt0() {
  asm volatile("s_wait_asynccnt 0x0" ::: "memory");
}
__device__ __forceinline__ unsigned lds_off(const void* p) {
  // LDS flat address: aperture base occupies bits [63:32]; low 32 bits are the
  // LDS byte offset, which is exactly what the async-LDS VDST operand wants.
  return (unsigned)(uintptr_t)p;
}

// ---------------------------------------------------------------------------
// f32 -> f16 conversion
// ---------------------------------------------------------------------------
__global__ void cvt_f32_to_f16(const float* __restrict__ in,
                               _Float16* __restrict__ out, size_t n) {
  size_t i = (size_t)blockIdx.x * blockDim.x + threadIdx.x;
  if (i < n) out[i] = (_Float16)in[i];
}

// ---------------------------------------------------------------------------
// Weight transpose + f32->f16:  W [K,N] row-major  ->  Wt [N,K] row-major f16
// LDS-tiled 32x32 so both global accesses are coalesced. gridDim.z = ensemble.
// ---------------------------------------------------------------------------
__global__ __launch_bounds__(256)
void transpose_cvt_f16(const float* __restrict__ W, _Float16* __restrict__ Wt,
                       int Kdim, int Ndim) {
  __shared__ float tile[32][33];
  const size_t ez = (size_t)blockIdx.z * (size_t)Kdim * Ndim;
  W  += ez;
  Wt += ez;
  const int k0 = blockIdx.x * 32;
  const int n0 = blockIdx.y * 32;
  const int tx = threadIdx.x & 31;
  const int ty = threadIdx.x >> 5;   // 0..7
  for (int i = ty; i < 32; i += 8)
    tile[i][tx] = W[(size_t)(k0 + i) * Ndim + n0 + tx];
  __syncthreads();
  for (int i = ty; i < 32; i += 8)
    Wt[(size_t)(n0 + i) * Kdim + k0 + tx] = (_Float16)tile[tx][i];
}

// ---------------------------------------------------------------------------
// WMMA GEMM: C[M,N] = act(A[M,K] * Bt[N,K]^T + bias) (+ residual)
//   A row-major f16, Bt = B^T row-major f16 (both K-contiguous).
//   256 threads = 8 waves as 2(M) x 4(N); each wave: 64x32 output =
//   4 M-frags x 2 N-frags = 8 WMMAs per 32-wide K-step.
//   Double-buffered LDS tiles filled by async-to-LDS copies.
//   gridDim.z = ensemble with element strides sA,sB,sBias,sC.
//   act: 0 = none, 1 = exact GELU.  Cf/Ch/residual/bias nullable.
// ---------------------------------------------------------------------------
__global__ __launch_bounds__(256)
void gemm_wmma_f16(const _Float16* __restrict__ A,
                   const _Float16* __restrict__ Bt,
                   const float* __restrict__ bias,
                   const float* __restrict__ residual,
                   float* __restrict__ Cf,
                   _Float16* __restrict__ Ch,
                   int M, int N, int K, int act,
                   size_t sA, size_t sB, size_t sBias, size_t sC) {
  const int e = blockIdx.z;
  A  += (size_t)e * sA;
  Bt += (size_t)e * sB;
  if (bias) bias += (size_t)e * sBias;
  if (Cf)   Cf   += (size_t)e * sC;
  if (Ch)   Ch   += (size_t)e * sC;

  __shared__ __align__(64) _Float16 As[2][GEMM_TM][GEMM_TK];  // 2 x 8 KB
  __shared__ __align__(64) _Float16 Bs[2][GEMM_TN][GEMM_TK];  // 2 x 8 KB

  const int tid  = threadIdx.x;
  const int lane = tid & 31;
  const int wave = tid >> 5;
  const int wm   = wave >> 2;         // 0..1 -> 64 rows each
  const int wn   = wave & 3;          // 0..3 -> 32 cols each
  const int m0   = blockIdx.y * GEMM_TM;
  const int n0   = blockIdx.x * GEMM_TN;
  const int l15  = lane & 15;
  const int h16  = lane >> 4;         // 0: lanes 0-15, 1: lanes 16-31

  // per-thread async-copy slot: 128 rows x 2 chunks of 16 halfs = 256 threads
  const int cr = tid >> 1;            // 0..127
  const int cc = (tid & 1) * 16;      // 0 or 16
  const _Float16* gA = A  + (size_t)(m0 + cr) * K + cc;
  const _Float16* gB = Bt + (size_t)(n0 + cr) * K + cc;

  v8f acc[4][2];
#pragma unroll
  for (int i = 0; i < 4; ++i) {
    acc[i][0] = (v8f){};
    acc[i][1] = (v8f){};
  }

  // prologue: tile 0 -> buffer 0
  async_copy_b128(lds_off(&As[0][cr][cc]), gA);
  async_copy_b128(lds_off(&Bs[0][cr][cc]), gB);
  wait_asynccnt0();
  __syncthreads();

  for (int k0 = 0; k0 < K; k0 += GEMM_TK) {
    const int cur = (k0 >> 5) & 1;
    if (k0 + GEMM_TK < K) {           // background-copy next tile
      async_copy_b128(lds_off(&As[cur ^ 1][cr][cc]), gA + k0 + GEMM_TK);
      async_copy_b128(lds_off(&Bs[cur ^ 1][cr][cc]), gB + k0 + GEMM_TK);
    }

    // A fragments (16x32): lane<16 holds K{0..15}, lane>=16 K{16..31}
    v16h af[4];
#pragma unroll
    for (int mf = 0; mf < 4; ++mf) {
      union { struct { u32x4 lo, hi; } s; v16h v; } fa;
      const _Float16* ap = &As[cur][wm * 64 + mf * 16 + l15][h16 * 16];
      fa.s.lo = *(const u32x4*)(ap);
      fa.s.hi = *(const u32x4*)(ap + 8);
      af[mf] = fa.v;
    }
    // B fragments (32x16): column = lane&15, 16 contiguous K per lane
    v16h bf[2];
#pragma unroll
    for (int nf = 0; nf < 2; ++nf)
      bf[nf] = *(const v16h*)&Bs[cur][wn * 32 + nf * 16 + l15][h16 * 16];

#pragma unroll
    for (int mf = 0; mf < 4; ++mf)
#pragma unroll
      for (int nf = 0; nf < 2; ++nf)
        acc[mf][nf] = __builtin_amdgcn_wmma_f32_16x16x32_f16(
            false, af[mf], false, bf[nf], (short)0, acc[mf][nf], false, false);

    wait_asynccnt0();
    __syncthreads();
  }

  // epilogue: C layout per tile — VGPR r: row = r + 8*h16, col = lane&15
#pragma unroll
  for (int mf = 0; mf < 4; ++mf) {
#pragma unroll
    for (int nf = 0; nf < 2; ++nf) {
      const int col = n0 + wn * 32 + nf * 16 + l15;
      const float bb = bias ? bias[col] : 0.0f;
#pragma unroll
      for (int r = 0; r < 8; ++r) {
        const int row = m0 + wm * 64 + mf * 16 + r + h16 * 8;
        float v = acc[mf][nf][r] + bb;
        if (act == 1)  // exact GELU (matches approximate=False)
          v = 0.5f * v * (1.0f + erff(v * 0.70710678118654752f));
        const size_t idx = (size_t)row * N + col;
        if (residual) v += residual[idx];
        if (Cf) Cf[idx] = v;
        if (Ch) Ch[idx] = (_Float16)v;
      }
    }
  }
}

// ---------------------------------------------------------------------------
// LRU diagonal complex recurrence: one thread per (b,d) channel.
//   h_t = lam * h_{t-1} + gamma*(in_r + i*in_i),  lam = exp(-exp(nu_log)+i*theta)
// Writes h.real/h.imag as f16 (WMMA A operands for middle_proj) and the final
// state [B, 2D] directly into the `hidden` output slot.
// ---------------------------------------------------------------------------
__global__ void lru_scan(const float* __restrict__ u,          // [3,B,T,D]
                         const float* __restrict__ params_log, // [3,D]
                         _Float16* __restrict__ hre,
                         _Float16* __restrict__ him,
                         float* __restrict__ hidden) {         // [B,2D]
  const int idx = blockIdx.x * blockDim.x + threadIdx.x;
  if (idx >= LRU_B * LRU_D) return;
  const int b = idx / LRU_D;
  const int d = idx % LRU_D;

  const float nu    = expf(params_log[d]);
  const float theta = expf(params_log[LRU_D + d]);
  const float gamma = expf(params_log[2 * LRU_D + d]);
  const float rmod  = expf(-nu);
  const float lre   = rmod * cosf(theta);
  const float lim   = rmod * sinf(theta);

  const size_t BTD = (size_t)LRU_B * LRU_T * LRU_D;
  const float* ur = u + (size_t)b * LRU_T * LRU_D + d;         // u[0] = in_r
  const float* ui = ur + BTD;                                  // u[1] = in_i
  _Float16* pr = hre + (size_t)b * LRU_T * LRU_D + d;
  _Float16* pi = him + (size_t)b * LRU_T * LRU_D + d;

  float hr = 0.0f, hi = 0.0f;
  for (int t = 0; t < LRU_T; ++t) {
    const size_t o = (size_t)t * LRU_D;
    const float xr = gamma * ur[o];
    const float xi = gamma * ui[o];
    const float nr = lre * hr - lim * hi + xr;
    const float ni = lre * hi + lim * hr + xi;
    hr = nr; hi = ni;
    pr[o] = (_Float16)hr;
    pi[o] = (_Float16)hi;
  }
  hidden[(size_t)b * 2 * LRU_D + d]         = hr;
  hidden[(size_t)b * 2 * LRU_D + LRU_D + d] = hi;
}

// ---------------------------------------------------------------------------
// out = mid0 - mid1 + o  (f32 result, plus f16 copy for FFN1 A-operand).
// outf may alias o (pure elementwise).
// ---------------------------------------------------------------------------
__global__ void combine_mid(const float* __restrict__ m0,
                            const float* __restrict__ m1,
                            const float* __restrict__ o,
                            float* __restrict__ outf,
                            _Float16* __restrict__ outh, size_t n) {
  size_t i = (size_t)blockIdx.x * blockDim.x + threadIdx.x;
  if (i < n) {
    float v = m0[i] - m1[i] + o[i];
    outf[i] = v;
    outh[i] = (_Float16)v;
  }
}

// ---------------------------------------------------------------------------
// In-place row LayerNorm over D=1024, one 256-thread block per row.
// ---------------------------------------------------------------------------
__global__ __launch_bounds__(256)
void layernorm_rows(float* __restrict__ y,
                    const float* __restrict__ g,
                    const float* __restrict__ b) {
  float* p = y + (size_t)blockIdx.x * LRU_D;
  __shared__ float s1[256];
  __shared__ float s2[256];
  float a = 0.0f, q = 0.0f;
  for (int i = threadIdx.x; i < LRU_D; i += 256) {
    float v = p[i];
    a += v; q += v * v;
  }
  s1[threadIdx.x] = a; s2[threadIdx.x] = q;
  __syncthreads();
  for (int s = 128; s > 0; s >>= 1) {
    if (threadIdx.x < s) {
      s1[threadIdx.x] += s1[threadIdx.x + s];
      s2[threadIdx.x] += s2[threadIdx.x + s];
    }
    __syncthreads();
  }
  const float mean = s1[0] * (1.0f / LRU_D);
  const float var  = s2[0] * (1.0f / LRU_D) - mean * mean;
  const float inv  = rsqrtf(var + 1e-5f);
  for (int i = threadIdx.x; i < LRU_D; i += 256)
    p[i] = (p[i] - mean) * inv * g[i] + b[i];
}

// ---------------------------------------------------------------------------
// Host-side orchestration
// ---------------------------------------------------------------------------
extern "C" void kernel_launch(void* const* d_in, const int* in_sizes, int n_in,
                              void* d_out, int out_size, void* d_ws, size_t ws_size,
                              hipStream_t stream) {
  const float* x          = (const float*)d_in[0];   // [B,T,D_IN]
  const float* W_in       = (const float*)d_in[1];   // [3,D_IN,D]
  const float* b_in       = (const float*)d_in[2];   // [3,D]
  const float* params_log = (const float*)d_in[3];   // [3,D]
  const float* W_mid      = (const float*)d_in[4];   // [2,D,D]
  const float* b_mid      = (const float*)d_in[5];   // [2,D]
  const float* w1_w       = (const float*)d_in[6];   // [D,D]
  const float* w1_b       = (const float*)d_in[7];   // [D]
  const float* w2_w       = (const float*)d_in[8];   // [D,D]
  const float* w2_b       = (const float*)d_in[9];   // [D]
  const float* ln_g       = (const float*)d_in[10];  // [D]
  const float* ln_b       = (const float*)d_in[11];  // [D]
  float* out = (float*)d_out;                        // y [B,T,D] ++ hidden [1,B,2D]

  const int    M  = LRU_B * LRU_T;                   // 32768 rows
  const int    D  = LRU_D;                           // 1024
  const size_t MN = (size_t)M * D;                   // 33,554,432

  // ---- workspace layout (with lifetime-based reuse) ----
  char* ws = (char*)d_ws;
  float*    u     = (float*)ws;                          // [3,M,D] f32 (in_r,in_i,o)
                                                         //   -> reused as mid0, mid1, out_f32
  _Float16* xh    = (_Float16*)(ws + 3 * MN * 4);        // [M,D] f16 x  -> reused as out_f16
  _Float16* hreh  = xh + MN;                             // [M,D] f16 h.re -> reused as gelu(t1)
  _Float16* himh  = hreh + MN;                           // [M,D] f16 h.im
  _Float16* winT  = himh + MN;                           // [3,D,D] f16  W_in^T  (N,K)
  _Float16* wmidT = winT + 3 * (size_t)D * D;            // [2,D,D] f16  W_mid^T
  _Float16* w1T   = wmidT + 2 * (size_t)D * D;           // [D,D] f16   w1^T
  _Float16* w2T   = w1T + (size_t)D * D;                 // [D,D] f16   w2^T

  const dim3 blk(256);

  // 0a) activation precision conversion (A operands stay row-major)
  cvt_f32_to_f16<<<dim3((unsigned)((MN + 255) / 256)), blk, 0, stream>>>(x, xh, MN);
  // 0b) weight transpose+convert: [K,N] f32 -> [N,K] f16
  transpose_cvt_f16<<<dim3(D / 32, D / 32, 3), blk, 0, stream>>>(W_in,  winT,  D, D);
  transpose_cvt_f16<<<dim3(D / 32, D / 32, 2), blk, 0, stream>>>(W_mid, wmidT, D, D);
  transpose_cvt_f16<<<dim3(D / 32, D / 32, 1), blk, 0, stream>>>(w1_w,  w1T,   D, D);
  transpose_cvt_f16<<<dim3(D / 32, D / 32, 1), blk, 0, stream>>>(w2_w,  w2T,   D, D);

  const dim3 gtile(D / GEMM_TN, M / GEMM_TM);            // (8, 256)

  // 1) in_proj: u[e] = x @ W_in[e] + b_in[e], e = 0..2
  gemm_wmma_f16<<<dim3(gtile.x, gtile.y, 3), blk, 0, stream>>>(
      xh, winT, b_in, nullptr, u, nullptr,
      M, D, D, /*act=*/0,
      /*sA=*/0, /*sB=*/(size_t)D * D, /*sBias=*/(size_t)D, /*sC=*/MN);

  // 2) diagonal complex scan; writes h (f16) and hidden -> d_out tail
  lru_scan<<<dim3((LRU_B * D + 255) / 256), blk, 0, stream>>>(
      u, params_log, hreh, himh, out + MN);

  // 3) middle_proj: mid[e] = h[e] @ W_mid[e] + b_mid[e] (h0=re, h1=im)
  //    in_r/in_i slots of `u` are dead now -> reuse for mid0/mid1
  gemm_wmma_f16<<<dim3(gtile.x, gtile.y, 2), blk, 0, stream>>>(
      hreh, wmidT, b_mid, nullptr, u, nullptr,
      M, D, D, /*act=*/0,
      /*sA=*/MN, /*sB=*/(size_t)D * D, /*sBias=*/(size_t)D, /*sC=*/MN);

  // 4) out = mid0 - mid1 + o ; keep f32 (residual) + f16 (FFN1 A)
  float* o_f32   = u + 2 * MN;    // in-place over the `o` slot
  float* out_f32 = o_f32;
  combine_mid<<<dim3((unsigned)((MN + 255) / 256)), blk, 0, stream>>>(
      u, u + MN, o_f32, out_f32, xh, MN);

  // 5) FFN1: t1 = gelu(out @ w1 + b1) -> f16 (reuse h.re slot)
  gemm_wmma_f16<<<gtile, blk, 0, stream>>>(
      xh, w1T, w1_b, nullptr, nullptr, hreh,
      M, D, D, /*act=*/1, 0, 0, 0, 0);

  // 6) FFN2: y_pre = t1 @ w2 + b2 + out  -> d_out
  gemm_wmma_f16<<<gtile, blk, 0, stream>>>(
      hreh, w2T, w2_b, out_f32, out, nullptr,
      M, D, D, /*act=*/0, 0, 0, 0, 0);

  // 7) LayerNorm in place on d_out rows
  layernorm_rows<<<dim3((unsigned)M), blk, 0, stream>>>(out, ln_g, ln_b);
}
